// Swin3DTransformerBlock_6605659701802
// MI455X (gfx1250) — matrix operand order repaired and checked by reading, not compiled, not run on hardware
//
#include <hip/hip_runtime.h>
#include <math.h>
#include <stdint.h>

typedef unsigned short u16;
typedef __attribute__((ext_vector_type(16))) __bf16 v16bf;
typedef __attribute__((ext_vector_type(8)))  float  v8f;
typedef __attribute__((ext_vector_type(4)))  unsigned int v4u;
typedef __attribute__((ext_vector_type(8)))  int v8i;
typedef __attribute__((ext_vector_type(4)))  int v4i;

#if defined(__HIP_DEVICE_COMPILE__) && __has_builtin(__builtin_amdgcn_tensor_load_to_lds) && __has_builtin(__builtin_amdgcn_s_wait_tensorcnt)
#define HAVE_TDM 1
#else
#define HAVE_TDM 0
#endif

union FragAB { uint4 q[2]; v16bf v; };

__device__ __forceinline__ u16 f2bf(float f) {
  union { float f; unsigned int u; } x; x.f = f;
  unsigned int u = x.u;
  unsigned int r = u + 0x7FFFu + ((u >> 16) & 1u);   // round-to-nearest-even
  return (u16)(r >> 16);
}

// window-row r -> token index l (fuses cyclic shift + window partition)
__device__ __forceinline__ int win_to_tok(int r) {
  int w = r / 144, t = r - w * 144;
  int ww_i = w & 7;            // W/12 = 8
  int wh_i = (w >> 3) & 15;    // H/6  = 16
  int wc_i = w >> 7;           // C/2  = 2
  int iw = t % 12; int ih = (t / 12) % 6; int ic = t / 72;
  int gc = wc_i * 2 + ic, gh = wh_i * 6 + ih, gw = ww_i * 12 + iw;
  int c = (gc + 1) & 3;
  int h = gh + 3; if (h >= 96) h -= 96;
  int ww = gw + 6; if (ww >= 96) ww -= 96;
  return (c * 96 + h) * 96 + ww;
}

// ---------------- conversion / permutation kernels ----------------

// x[token order, f32] -> xw[window order, bf16]
__global__ __launch_bounds__(256) void k_shiftwin(const float* __restrict__ x,
                                                  u16* __restrict__ xw) {
  int idx = blockIdx.x * 256 + threadIdx.x;   // 36864*128 threads, 4 elems each
  int r = idx >> 7, c4 = (idx & 127) * 4;
  int l = win_to_tok(r);
  float4 v = *(const float4*)(x + (size_t)l * 512 + c4);
  u16 o[4] = { f2bf(v.x), f2bf(v.y), f2bf(v.z), f2bf(v.w) };
  *(uint2*)(xw + (size_t)r * 512 + c4) = *(const uint2*)o;
}

// W[K,N] f32 -> Wt[N,K] bf16 (32x32 LDS tile transpose). Makes every GEMM
// operand row-major-contiguous so the TDM can stage it as a plain 2-D tile.
__global__ __launch_bounds__(256) void k_transpose_f2bf(const float* __restrict__ w,
                                                        u16* __restrict__ wt,
                                                        int K, int N) {
  __shared__ float t[32][33];
  int kb = blockIdx.y * 32, nb = blockIdx.x * 32;
  int tx = threadIdx.x & 31, ty = threadIdx.x >> 5;   // 32 x 8
#pragma unroll
  for (int i = 0; i < 32; i += 8)
    t[ty + i][tx] = w[(size_t)(kb + ty + i) * N + nb + tx];
  __syncthreads();
#pragma unroll
  for (int i = 0; i < 32; i += 8)
    wt[(size_t)(nb + ty + i) * K + kb + tx] = f2bf(t[tx][ty + i]);
}

// mod = silu(c) @ W + b   (tiny GEMV: T=512 -> O=1024)
__global__ __launch_bounds__(256) void k_mod(const float* __restrict__ c,
                                             const float* __restrict__ w,
                                             const float* __restrict__ b,
                                             float* __restrict__ mod,
                                             int T, int O) {
  int j = blockIdx.x * 256 + threadIdx.x;
  if (j >= O) return;
  float s = 0.f;
  for (int t = 0; t < T; ++t) {
    float cv = c[t];
    s += (cv / (1.f + __expf(-cv))) * w[t * O + j];
  }
  mod[j] = s + b[j];
}

// ---------------- bf16 WMMA GEMM with TDM-staged tiles ----------------
// C[M,N] = act(A[M,K] @ Bt[N,K]^T + bias). A, Bt bf16 row-major, f32 acc.
// Tiles: BM=BN=128, BK=32; 8 waves, each wave 32x64 = 2x4 wmma fragments.
#define BM 128
#define BN 128
#define BK 32
#define LDT 40   // LDS row pitch (bf16 elems): 32 data + 4-DWORD TDM pad

// Issue one TDM 2-D tile load: tile 32(K) x 128(rows), element 2B, with
// pad_enable so each 64B row gets +16B -> LDS pitch of 80B (= LDT elems).
// pad_interval=3 (16 DWORDs = 64B), pad_amount=3 (4 DWORDs = 16B).
__device__ __forceinline__ void tdm_tile_load(const u16* gtile, unsigned lds_off,
                                              int K, int rows) {
#if HAVE_TDM
  unsigned long long ga = (unsigned long long)(uintptr_t)gtile;
  v4u g0 = { 1u,                                   // count=1, user descriptor
             lds_off,                              // lds_addr
             (unsigned)ga,                         // global_addr[31:0]
             (unsigned)(ga >> 32) | (2u << 30) };  // global_addr[56:32] | type=2
  v8i g1;
  g1[0] = (int)((1u << 16) | (1u << 20) | (3u << 22) | (3u << 25)); // data_size=2B, pad_en, pi=3, pa=3
  g1[1] = (int)(((unsigned)K & 0xFFFFu) << 16);                     // tensor_dim0[15:0]
  g1[2] = (int)(((unsigned)K >> 16) | (((unsigned)rows & 0xFFFFu) << 16)); // dim0 hi | dim1 lo
  g1[3] = (int)((((unsigned)rows >> 16) & 0xFFFFu) | ((unsigned)BK << 16)); // dim1 hi | tile_dim0
  g1[4] = BM;                                       // tile_dim1=128, tile_dim2=0
  g1[5] = K;                                        // tensor_dim0_stride[31:0]
  g1[6] = 0;
  g1[7] = 0;
  v4i z4 = { 0, 0, 0, 0 };
  v8i z8 = { 0, 0, 0, 0, 0, 0, 0, 0 };
  __builtin_amdgcn_tensor_load_to_lds(g0, g1, z4, z4, z8, 0);
#else
  (void)gtile; (void)lds_off; (void)K; (void)rows;
#endif
}

template <bool GELU, bool WF, bool WB>
__global__ __launch_bounds__(256) void k_gemm(const u16* __restrict__ A,
                                              const u16* __restrict__ Bt,
                                              const float* __restrict__ bias,
                                              float* __restrict__ Cf,
                                              u16* __restrict__ Cb,
                                              int M, int N, int K) {
  __shared__ u16 As[2][BM * LDT];
  __shared__ u16 Bs[2][BN * LDT];
  int tid = threadIdx.x, lane = tid & 31, wv = tid >> 5;
  int wm = wv & 3, wn = wv >> 2;
  int m0 = blockIdx.y * BM, n0 = blockIdx.x * BN;
  int half = lane >> 4, l15 = lane & 15;

  v8f zero = {};
  v8f acc[2][4];
#pragma unroll
  for (int i = 0; i < 2; ++i)
#pragma unroll
    for (int j = 0; j < 4; ++j) acc[i][j] = zero;

  const u16* Ag = A + (size_t)m0 * K;
  const u16* Bg = Bt + (size_t)n0 * K;
  int nK = K / BK;

#if HAVE_TDM
  unsigned aoff = (unsigned)(uintptr_t)(&As[0][0]);
  unsigned boff = (unsigned)(uintptr_t)(&Bs[0][0]);
  if (wv == 0) {                       // one wave drives the TDM
    tdm_tile_load(Ag, aoff, K, M);
    tdm_tile_load(Bg, boff, K, N);
  }
#endif

  for (int kt = 0; kt < nK; ++kt) {
#if HAVE_TDM
    int buf = kt & 1;
    if (wv == 0) {
      if (kt + 1 < nK) {               // prefetch next pair into other buffer
        unsigned bo = (unsigned)((buf ^ 1) * BM * LDT * 2);
        tdm_tile_load(Ag + (size_t)(kt + 1) * BK, aoff + bo, K, M);
        tdm_tile_load(Bg + (size_t)(kt + 1) * BK, boff + bo, K, N);
        __builtin_amdgcn_s_wait_tensorcnt(2);   // oldest pair (kt) complete
      } else {
        __builtin_amdgcn_s_wait_tensorcnt(0);
      }
    }
    __syncthreads();
#else
    int buf = 0;
#pragma unroll
    for (int i = 0; i < 2; ++i) {      // both operands contiguous: 16B copies
      int idx = i * 256 + tid;
      int row = idx >> 2, kc = (idx & 3) * 8;
      *(uint4*)(&As[0][0] + row * LDT + kc) =
          *(const uint4*)(Ag + (size_t)row * K + kt * BK + kc);
      *(uint4*)(&Bs[0][0] + row * LDT + kc) =
          *(const uint4*)(Bg + (size_t)row * K + kt * BK + kc);
    }
    __syncthreads();
#endif
    const u16* Asb = &As[buf][0];
    const u16* Bsb = &Bs[buf][0];

    FragAB a[2], b[4];
#pragma unroll
    for (int i = 0; i < 2; ++i) {      // A 16x32: VGPR0-3 K=half*8.., VGPR4-7 K=16+half*8..
      const u16* p = Asb + (wm * 32 + i * 16 + l15) * LDT + half * 8;
      a[i].q[0] = *(const uint4*)p;
      a[i].q[1] = *(const uint4*)(p + 16);
    }
#pragma unroll
    for (int j = 0; j < 4; ++j) {      // B 32x16: lane=col, VGPR0-3 K=half*8.., VGPR4-7 K=16+half*8..
      const u16* p = Bsb + (wn * 64 + j * 16 + l15) * LDT + half * 8;
      b[j].q[0] = *(const uint4*)p;
      b[j].q[1] = *(const uint4*)(p + 16);
    }
#pragma unroll
    for (int i = 0; i < 2; ++i)
#pragma unroll
      for (int j = 0; j < 4; ++j)
        acc[i][j] = __builtin_amdgcn_wmma_f32_16x16x32_bf16(
            false, a[i].v, false, b[j].v, (short)0, acc[i][j], false, false);
    __syncthreads();
  }

  // epilogue (compile-time specialized: no runtime branches)
#pragma unroll
  for (int j = 0; j < 4; ++j) {
    int col = n0 + wn * 64 + j * 16 + l15;
    float bv = bias[col];
#pragma unroll
    for (int i = 0; i < 2; ++i) {
      int rbase = m0 + wm * 32 + i * 16 + half * 8;
#pragma unroll
      for (int v = 0; v < 8; ++v) {
        float val = acc[i][j][v] + bv;
        if (GELU) val = 0.5f * val * (1.f + erff(val * 0.70710678118654752f));
        size_t off = (size_t)(rbase + v) * N + col;
        if (WF) Cf[off] = val;
        if (WB) Cb[off] = f2bf(val);
      }
    }
  }
}

// ---------------- windowed attention (one WG per window) ----------------
__global__ __launch_bounds__(256) void k_attn(const u16* __restrict__ qkv,
                                              u16* __restrict__ out) {
  extern __shared__ char smem[];
  int* labels = (int*)smem;                       //  640 B (144 used)
  u16* qs = (u16*)(smem + 640);                   // 144 x 40   (11520 B)
  u16* ks = (u16*)(smem + 640 + 11520);           // 144 x 40   (11520 B)
  u16* vt = (u16*)(smem + 640 + 23040);           //  32 x 160  (10240 B) v^T, K padded
  float* P = (float*)(smem + 640 + 33280);        // 144 x 160 f32 (92160 B)
  u16* pb = (u16*)P;                              // bf16 alias, row stride 320 u16

  int tid = threadIdx.x, lane = tid & 31, wv = tid >> 5;
  int half = lane >> 4, l15 = lane & 15;
  int w = blockIdx.x;
  int ww_i = w & 7, wh_i = (w >> 3) & 15, wc_i = w >> 7;
  const float scale = 0.17677669529663687f;       // 1/sqrt(32)

  for (int t = tid; t < 144; t += 256) {          // Swin region labels
    int iw = t % 12, ih = (t / 12) % 6, ic = t / 72;
    int gc = wc_i * 2 + ic, gh = wh_i * 6 + ih, gw = ww_i * 12 + iw;
    int bc = gc < 2 ? 0 : (gc < 3 ? 1 : 2);
    int bh = gh < 90 ? 0 : (gh < 93 ? 1 : 2);
    int bw = gw < 84 ? 0 : (gw < 90 ? 1 : 2);
    labels[t] = (bc * 3 + bh) * 3 + bw;
  }
  for (int i = tid; i < 32 * 16; i += 256) vt[(i >> 4) * 160 + 144 + (i & 15)] = 0;

  size_t rowbase = (size_t)w * 144;

  for (int h = 0; h < 16; ++h) {
    __syncthreads();
    for (int idx = tid; idx < 3456; idx += 256) { // stage q, k, v^T
      int mat = idx / 1152, rem = idx - mat * 1152;
      int t = rem >> 3, c4 = (rem & 7) * 4;
      uint2 d = *(const uint2*)(qkv + (rowbase + t) * 1536 + mat * 512 + h * 32 + c4);
      if (mat == 0) {
        *(uint2*)(qs + t * 40 + c4) = d;
      } else if (mat == 1) {
        *(uint2*)(ks + t * 40 + c4) = d;
      } else {
        const u16* e = (const u16*)&d;
        vt[(c4 + 0) * 160 + t] = e[0];
        vt[(c4 + 1) * 160 + t] = e[1];
        vt[(c4 + 2) * 160 + t] = e[2];
        vt[(c4 + 3) * 160 + t] = e[3];
      }
    }
    __syncthreads();
    for (int tile = wv; tile < 81; tile += 8) {   // scores = q @ k^T (9x9 tiles)
      int ti = tile / 9, tj = tile - ti * 9;
      FragAB a, b;
      const u16* pa = qs + (ti * 16 + l15) * 40 + half * 8;
      a.q[0] = *(const uint4*)pa; a.q[1] = *(const uint4*)(pa + 16);
      const u16* pk = ks + (tj * 16 + l15) * 40 + half * 8;
      b.q[0] = *(const uint4*)pk; b.q[1] = *(const uint4*)(pk + 16);
      v8f c = {};
      c = __builtin_amdgcn_wmma_f32_16x16x32_bf16(false, a.v, false, b.v,
                                                  (short)0, c, false, false);
      int col = tj * 16 + l15, rb = ti * 16 + half * 8;
#pragma unroll
      for (int v = 0; v < 8; ++v) P[(rb + v) * 160 + col] = c[v];
    }
    __syncthreads();
    if (tid < 144) {                              // masked softmax per row
      int r = tid, lr = labels[r];
      float mx = -1e30f;
      for (int j = 0; j < 144; ++j) {
        float s = P[r * 160 + j] * scale + (labels[j] == lr ? 0.f : -100.f);
        mx = fmaxf(mx, s);
      }
      float sum = 0.f;
      for (int j = 0; j < 144; ++j) {
        float s = P[r * 160 + j] * scale + (labels[j] == lr ? 0.f : -100.f);
        float e = __expf(s - mx);
        P[r * 160 + j] = e;
        sum += e;
      }
      float inv = 1.f / sum;
      for (int j = 0; j < 144; ++j) {
        float e = P[r * 160 + j];                 // read f32 before bf16 overwrite
        pb[r * 320 + j] = f2bf(e * inv);
      }
      for (int j = 144; j < 160; ++j) pb[r * 320 + j] = 0;
    }
    __syncthreads();
    for (int tile = wv; tile < 18; tile += 8) {   // out = P @ v (9x2 tiles, K=160)
      int ti = tile >> 1, tj = tile & 1;
      v8f c = {};
#pragma unroll
      for (int kk = 0; kk < 5; ++kk) {
        FragAB a, b;
        const u16* pa = pb + (ti * 16 + l15) * 320 + kk * 32 + half * 8;
        a.q[0] = *(const uint4*)pa; a.q[1] = *(const uint4*)(pa + 16);
        const u16* pv = vt + (tj * 16 + l15) * 160 + kk * 32 + half * 8;
        b.q[0] = *(const uint4*)pv; b.q[1] = *(const uint4*)(pv + 16);
        c = __builtin_amdgcn_wmma_f32_16x16x32_bf16(false, a.v, false, b.v,
                                                    (short)0, c, false, false);
      }
      int col = h * 32 + tj * 16 + l15, rb = ti * 16 + half * 8;
#pragma unroll
      for (int v = 0; v < 8; ++v)
        out[(rowbase + rb + v) * 512 + col] = f2bf(c[v]);
    }
  }
}

// ---------------- AdaLN + residual (optional window->token scatter) ----------
__global__ __launch_bounds__(256) void k_adaln(const float* __restrict__ src,
                                               const float* __restrict__ resid,
                                               const float* __restrict__ mod,
                                               float* __restrict__ dstf,
                                               u16* __restrict__ dstb,
                                               int permute) {
  __shared__ float red[256];
  __shared__ float s_mean, s_rstd;
  int r = blockIdx.x, tid = threadIdx.x;
  const float* row = src + (size_t)r * 512;
  float v0 = row[tid], v1 = row[tid + 256];
  red[tid] = v0 + v1;
  __syncthreads();
  for (int s = 128; s > 0; s >>= 1) {
    if (tid < s) red[tid] += red[tid + s];
    __syncthreads();
  }
  if (tid == 0) s_mean = red[0] * (1.f / 512.f);
  __syncthreads();
  float m = s_mean;
  float d0 = v0 - m, d1 = v1 - m;
  red[tid] = d0 * d0 + d1 * d1;
  __syncthreads();
  for (int s = 128; s > 0; s >>= 1) {
    if (tid < s) red[tid] += red[tid + s];
    __syncthreads();
  }
  if (tid == 0) s_rstd = rsqrtf(red[0] * (1.f / 512.f) + 1e-5f);
  __syncthreads();
  float rs = s_rstd;
  int l = permute ? win_to_tok(r) : r;
#pragma unroll
  for (int e = 0; e < 2; ++e) {
    int j = tid + e * 256;
    float val = (e == 0 ? d0 : d1) * rs;
    float y = val * mod[512 + j] + mod[j];        // SCALE_BIAS = 0
    float o = resid[(size_t)l * 512 + j] + y;
    if (dstf) dstf[(size_t)l * 512 + j] = o;
    if (dstb) dstb[(size_t)l * 512 + j] = f2bf(o);
  }
}

// ---------------- launch ----------------

extern "C" void kernel_launch(void* const* d_in, const int* in_sizes, int n_in,
                              void* d_out, int out_size, void* d_ws, size_t ws_size,
                              hipStream_t stream) {
  (void)in_sizes; (void)n_in; (void)out_size; (void)ws_size;
  const float* x      = (const float*)d_in[0];
  const float* c      = (const float*)d_in[1];
  const float* w_qkv  = (const float*)d_in[2];
  const float* b_qkv  = (const float*)d_in[3];
  const float* w_proj = (const float*)d_in[4];
  const float* b_proj = (const float*)d_in[5];
  const float* w_fc1  = (const float*)d_in[6];
  const float* b_fc1  = (const float*)d_in[7];
  const float* w_fc2  = (const float*)d_in[8];
  const float* b_fc2  = (const float*)d_in[9];
  const float* w_mod1 = (const float*)d_in[10];
  const float* b_mod1 = (const float*)d_in[11];
  const float* w_mod2 = (const float*)d_in[12];
  const float* b_mod2 = (const float*)d_in[13];

  const int L = 36864, D = 512, NQ = 1536, HD = 2048;

  char* ws = (char*)d_ws;
  size_t o = 0;
  auto take = [&](size_t bytes) { char* p = ws + o; o = (o + bytes + 255) & ~(size_t)255; return p; };
  u16*   wqkvT  = (u16*)take((size_t)D * NQ * 2);   // [NQ, D]
  u16*   wprojT = (u16*)take((size_t)D * D * 2);    // [D, D]
  u16*   wfc1T  = (u16*)take((size_t)D * HD * 2);   // [HD, D]
  u16*   wfc2T  = (u16*)take((size_t)HD * D * 2);   // [D, HD]
  u16*   xw     = (u16*)take((size_t)L * D * 2);
  u16*   qkvb   = (u16*)take((size_t)L * NQ * 2);
  u16*   attnb  = (u16*)take((size_t)L * D * 2);
  float* projf  = (float*)take((size_t)L * D * 4);
  float* x1     = (float*)take((size_t)L * D * 4);
  u16*   x1b    = (u16*)take((size_t)L * D * 2);
  u16*   hb     = (u16*)take((size_t)L * HD * 2);
  float* h2     = (float*)take((size_t)L * D * 4);
  float* mod1   = (float*)take(1024 * 4);
  float* mod2   = (float*)take(1024 * 4);

  // weight transpose + bf16 downconvert (one-time, makes tiles TDM-friendly)
  k_transpose_f2bf<<<dim3(NQ / 32, D / 32), 256, 0, stream>>>(w_qkv, wqkvT, D, NQ);
  k_transpose_f2bf<<<dim3(D / 32, D / 32), 256, 0, stream>>>(w_proj, wprojT, D, D);
  k_transpose_f2bf<<<dim3(HD / 32, D / 32), 256, 0, stream>>>(w_fc1, wfc1T, D, HD);
  k_transpose_f2bf<<<dim3(D / 32, HD / 32), 256, 0, stream>>>(w_fc2, wfc2T, HD, D);

  // shift + window-partition + bf16 gather
  k_shiftwin<<<(L * 128) / 256, 256, 0, stream>>>(x, xw);

  // AdaLN modulation vectors
  k_mod<<<4, 256, 0, stream>>>(c, w_mod1, b_mod1, mod1, 512, 1024);
  k_mod<<<4, 256, 0, stream>>>(c, w_mod2, b_mod2, mod2, 512, 1024);

  // QKV projection (bf16 out feeds attention)
  k_gemm<false, false, true><<<dim3(NQ / BN, L / BM), 256, 0, stream>>>(
      xw, wqkvT, b_qkv, nullptr, qkvb, L, NQ, D);

  // windowed attention (126080 B dynamic LDS per WG; CDNA5 allows 320 KB)
  k_attn<<<256, 256, 126080, stream>>>(qkvb, attnb);

  // output projection
  k_gemm<false, true, false><<<dim3(D / BN, L / BM), 256, 0, stream>>>(
      attnb, wprojT, b_proj, projf, nullptr, L, D, D);

  // x1 = x + adaLN(window_reverse(proj))
  k_adaln<<<L, 256, 0, stream>>>(projf, x, mod1, x1, x1b, 1);

  // MLP: fc1 + exact GELU (bf16 out), then fc2
  k_gemm<true, false, true><<<dim3(HD / BN, L / BM), 256, 0, stream>>>(
      x1b, wfc1T, b_fc1, nullptr, hb, L, HD, D);
  k_gemm<false, true, false><<<dim3(D / BN, L / BM), 256, 0, stream>>>(
      hb, wfc2T, b_fc2, h2, nullptr, L, D, HD);

  // out = x1 + adaLN(h2)
  k_adaln<<<L, 256, 0, stream>>>(h2, x1, mod2, (float*)d_out, nullptr, 0);
}
